// RNNX_12060268167652
// MI455X (gfx1250) — matrix operand order repaired and checked by reading, not compiled, not run on hardware
//
#include <hip/hip_runtime.h>
#include <math.h>

// Problem constants (from reference): T=512, B=128, I=256, H=512
#define T_   512
#define B_   128
#define I_   256
#define H_   512
#define NBLK 32           // persistent grid: 32 blocks x 256 threads = 256 waves (one 16x16 tile each)
#define KC1  24           // (I+H)/32  k-chunks for layer 1
#define KC2  32           // (H+H)/32  k-chunks for layer 2

typedef __attribute__((ext_vector_type(16))) _Float16 v16h;
typedef __attribute__((ext_vector_type(8)))  _Float16 v8h;
typedef __attribute__((ext_vector_type(8)))  float    v8f;
typedef __attribute__((ext_vector_type(4)))  float    v4f;

union Frag { v16h v; v8h h[2]; };

// ---------------------------------------------------------------------------
// Device-wide barrier (atomic counter, agent scope). Counter reset each launch.
// ---------------------------------------------------------------------------
__device__ __forceinline__ void grid_sync(unsigned* __restrict__ bar, unsigned& phase) {
    phase += 1;
    __threadfence();                       // release: flush this thread's stores to device scope
    __syncthreads();
    if (threadIdx.x == 0) {
        __hip_atomic_fetch_add(bar, 1u, __ATOMIC_RELEASE, __HIP_MEMORY_SCOPE_AGENT);
        const unsigned tgt = phase * (unsigned)NBLK;
        while (__hip_atomic_load(bar, __ATOMIC_ACQUIRE, __HIP_MEMORY_SCOPE_AGENT) < tgt) {
            __builtin_amdgcn_s_sleep(2);
        }
    }
    __syncthreads();
    __threadfence();                       // acquire: invalidate stale cached lines
}

// ---------------------------------------------------------------------------
// One WMMA step: A fragment from a row-major f16 row pointer (contiguous K
// order k = 16*(lane/16) + s), B fragment from the pre-packed weight buffer.
// ---------------------------------------------------------------------------
__device__ __forceinline__ v8f wmma_step(const _Float16* __restrict__ arow, int koff,
                                         const _Float16* __restrict__ fb, int lane, v8f acc) {
    Frag a, b;
    a.h[0] = *(const v8h*)(arow + 2 * koff);      // K j..j+7   (j = 0 or 16 by lane half)
    a.h[1] = *(const v8h*)(arow + 8 + 2 * koff);  // K j+8..j+15
    b.h[0] = *(const v8h*)(fb + lane * 8);        // packed: lane-major, v[0:3]
    b.h[1] = *(const v8h*)(fb + 256 + lane * 8);  // packed: v[4:7]
    return __builtin_amdgcn_wmma_f32_16x16x32_f16(false, a.v, false, b.v, (short)0, acc,
                                                  false, false);
}

// ---------------------------------------------------------------------------
// Persistent RNN kernel: 512 timesteps, 2 layers, 2 grid syncs per step.
// ---------------------------------------------------------------------------
__global__ __launch_bounds__(256) void rnn_persistent(
    const _Float16* __restrict__ X16,   // [T,B,I] f16
    const _Float16* __restrict__ W1p,   // packed B-fragments, 768 frags * 512 f16
    const _Float16* __restrict__ W2p,   // packed B-fragments, 1024 frags * 512 f16
    const float* __restrict__ b1, const float* __restrict__ b2,
    float* __restrict__ out,            // [T,B,H] then [2,B,H]
    _Float16* __restrict__ h1a, _Float16* __restrict__ h1b,
    _Float16* __restrict__ h2a, _Float16* __restrict__ h2b,
    unsigned* __restrict__ bar)
{
    const int lane  = threadIdx.x & 31;
    const int wave  = (blockIdx.x << 3) | (threadIdx.x >> 5);  // 0..255
    const int mtile = wave >> 5;                               // 0..7  (batch tiles)
    const int ntile = wave & 31;                               // 0..31 (H tiles)
    const int row   = mtile * 16 + (lane & 15);                // A-fragment batch row
    const int col   = ntile * 16 + (lane & 15);                // C/D column
    const int r0    = mtile * 16 + ((lane >> 4) << 3);         // C/D row base for acc[i]
    const int koff  = (lane < 16) ? 0 : 8;                     // A layout lane-half K offset
    const float bias1 = b1[col];
    const float bias2 = b2[col];
    const size_t OUTOFF = (size_t)T_ * B_ * H_;
    unsigned phase = 0;

    for (int t = 0; t < T_; ++t) {
        const _Float16* h1rd = (t & 1) ? h1b : h1a;
        _Float16*       h1wr = (t & 1) ? h1a : h1b;
        const _Float16* h2rd = (t & 1) ? h2b : h2a;
        _Float16*       h2wr = (t & 1) ? h2a : h2b;

        // ---------------- Layer 1: y1 = tanh([x_t, h1] @ W1 + b1) ----------------
        v8f acc = {};
        const _Float16* xrow = X16 + (size_t)t * (B_ * I_) + (size_t)row * I_;
        #pragma unroll 4
        for (int kc = 0; kc < 8; ++kc)       // K 0..255 from x_t
            acc = wmma_step(xrow + kc * 32, koff,
                            W1p + ((size_t)(ntile * KC1 + kc) << 9), lane, acc);
        const _Float16* h1row = h1rd + (size_t)row * H_;
        #pragma unroll 4
        for (int kc = 0; kc < 16; ++kc)      // K 256..767 from h1
            acc = wmma_step(h1row + kc * 32, koff,
                            W1p + ((size_t)(ntile * KC1 + 8 + kc) << 9), lane, acc);

        float y[8];
        #pragma unroll
        for (int i = 0; i < 8; ++i) y[i] = tanhf(acc[i] + bias1);
        #pragma unroll
        for (int i = 0; i < 8; ++i)
            h1wr[(size_t)(r0 + i) * H_ + col] = (_Float16)y[i];
        if (t == T_ - 1) {                   // final hidden state, layer 1
            #pragma unroll
            for (int i = 0; i < 8; ++i)
                out[OUTOFF + (size_t)(r0 + i) * H_ + col] = y[i];
        }

        grid_sync(bar, phase);               // y1 visible to all layer-2 tiles

        // ---------------- Layer 2: y2 = tanh([y1, h2] @ W2 + b2) ----------------
        acc = (v8f){};
        const _Float16* y1row = h1wr + (size_t)row * H_;
        #pragma unroll 4
        for (int kc = 0; kc < 16; ++kc)      // K 0..511 from y1(t)
            acc = wmma_step(y1row + kc * 32, koff,
                            W2p + ((size_t)(ntile * KC2 + kc) << 9), lane, acc);
        const _Float16* h2row = h2rd + (size_t)row * H_;
        #pragma unroll 4
        for (int kc = 0; kc < 16; ++kc)      // K 512..1023 from h2(t-1)
            acc = wmma_step(h2row + kc * 32, koff,
                            W2p + ((size_t)(ntile * KC2 + 16 + kc) << 9), lane, acc);

        #pragma unroll
        for (int i = 0; i < 8; ++i) y[i] = tanhf(acc[i] + bias2);
        float* orow = out + (size_t)t * (B_ * H_);
        #pragma unroll
        for (int i = 0; i < 8; ++i) {
            h2wr[(size_t)(r0 + i) * H_ + col] = (_Float16)y[i];
            orow[(size_t)(r0 + i) * H_ + col] = y[i];
        }
        if (t == T_ - 1) {                   // final hidden state, layer 2
            #pragma unroll
            for (int i = 0; i < 8; ++i)
                out[OUTOFF + (size_t)B_ * H_ + (size_t)(r0 + i) * H_ + col] = y[i];
        }

        grid_sync(bar, phase);               // WAR protection for next step's h buffers
    }
}

// ---------------------------------------------------------------------------
// Prep kernels (run once per launch; amortized over 512 steps)
// ---------------------------------------------------------------------------
__global__ void init_state(unsigned* bar, _Float16* h1a, _Float16* h2a) {
    const int i = blockIdx.x * blockDim.x + threadIdx.x;   // 65536 threads
    if (i == 0) *bar = 0u;
    if (i < B_ * H_) { h1a[i] = (_Float16)0.f; h2a[i] = (_Float16)0.f; }
}

__global__ void cvt_x(const float* __restrict__ X, _Float16* __restrict__ X16) {
    const int n4 = (T_ * B_ * I_) / 4;
    const int stride = gridDim.x * blockDim.x;
    for (int i = blockIdx.x * blockDim.x + threadIdx.x; i < n4; i += stride) {
        v4f x = *(const v4f*)(X + (size_t)i * 4);
        _Float16* o = X16 + (size_t)i * 4;
        o[0] = (_Float16)x.x; o[1] = (_Float16)x.y;
        o[2] = (_Float16)x.z; o[3] = (_Float16)x.w;
    }
}

// Pack W (row-major [K, 512] f32) into per-fragment lane-major f16 layout:
// fragment (ntile, kc) = 512 f16; element [half*256 + l*8 + j] =
//   W[kc*32 + half*8 + j + (l<16?0:16)][ntile*16 + (l&15)]
__global__ void pack_w(const float* __restrict__ W1, const float* __restrict__ W2,
                       _Float16* __restrict__ W1p, _Float16* __restrict__ W2p) {
    const int n1 = (I_ + H_) * H_;          // 393216
    const int n2 = (H_ + H_) * H_;          // 524288
    const int stride = gridDim.x * blockDim.x;
    for (int i = blockIdx.x * blockDim.x + threadIdx.x; i < n1 + n2; i += stride) {
        if (i < n1) {
            const int frag = i >> 9, w = i & 511;
            const int nt = frag / KC1, kc = frag % KC1;
            const int hf = w >> 8, rem = w & 255, l = rem >> 3, j = rem & 7;
            const int K = kc * 32 + hf * 8 + j + ((l < 16) ? 0 : 16);
            const int N = nt * 16 + (l & 15);
            W1p[i] = (_Float16)W1[(size_t)K * H_ + N];
        } else {
            const int i2 = i - n1;
            const int frag = i2 >> 9, w = i2 & 511;
            const int nt = frag >> 5, kc = frag & 31;
            const int hf = w >> 8, rem = w & 255, l = rem >> 3, j = rem & 7;
            const int K = kc * 32 + hf * 8 + j + ((l < 16) ? 0 : 16);
            const int N = nt * 16 + (l & 15);
            W2p[i2] = (_Float16)W2[(size_t)K * H_ + N];
        }
    }
}

// ---------------------------------------------------------------------------
extern "C" void kernel_launch(void* const* d_in, const int* in_sizes, int n_in,
                              void* d_out, int out_size, void* d_ws, size_t ws_size,
                              hipStream_t stream) {
    const float* Xt = (const float*)d_in[0];
    const float* W1 = (const float*)d_in[1];
    const float* b1 = (const float*)d_in[2];
    const float* W2 = (const float*)d_in[3];
    const float* b2 = (const float*)d_in[4];
    float* out = (float*)d_out;

    // Workspace carve (16B+ aligned blocks)
    char* ws = (char*)d_ws;
    unsigned* bar = (unsigned*)ws;                       size_t off = 256;
    _Float16* W1p = (_Float16*)(ws + off);               off += (size_t)(I_ + H_) * H_ * 2;  // 768 KiB
    _Float16* W2p = (_Float16*)(ws + off);               off += (size_t)(H_ + H_) * H_ * 2;  // 1 MiB
    _Float16* h1a = (_Float16*)(ws + off);               off += (size_t)B_ * H_ * 2;
    _Float16* h1b = (_Float16*)(ws + off);               off += (size_t)B_ * H_ * 2;
    _Float16* h2a = (_Float16*)(ws + off);               off += (size_t)B_ * H_ * 2;
    _Float16* h2b = (_Float16*)(ws + off);               off += (size_t)B_ * H_ * 2;
    _Float16* X16 = (_Float16*)(ws + off);               // 32 MiB

    init_state<<<256, 256, 0, stream>>>(bar, h1a, h2a);
    cvt_x<<<2048, 256, 0, stream>>>(Xt, X16);
    pack_w<<<1024, 256, 0, stream>>>(W1, W2, W1p, W2p);
    rnn_persistent<<<NBLK, 256, 0, stream>>>(X16, W1p, W2p, b1, b2, out,
                                             h1a, h1b, h2a, h2b, bar);
}
